// CSPLayer_71657234367118
// MI455X (gfx1250) — matrix operand order repaired and checked by reading, not compiled, not run on hardware
//
#include <hip/hip_runtime.h>
#include <hip/hip_bf16.h>

#define HIDDEN 128
#define EDGE_IN 268
#define K1PAD 288          // EDGE_IN padded up to multiple of 32 for 16x16x32 WMMA
#define LN_EPS 1e-5f

typedef __attribute__((ext_vector_type(4)))  unsigned v4u;
typedef __attribute__((ext_vector_type(8)))  unsigned v8u;
typedef __attribute__((ext_vector_type(16))) __bf16   v16bf;
typedef __attribute__((ext_vector_type(8)))  float    v8f;

__device__ __forceinline__ unsigned short f2bf(float f) {
    unsigned u = __float_as_uint(f);
    u += 0x7FFFu + ((u >> 16) & 1u);      // round-to-nearest-even
    return (unsigned short)(u >> 16);
}
__device__ __forceinline__ unsigned pack2(float lo, float hi) {
    return (unsigned)f2bf(lo) | ((unsigned)f2bf(hi) << 16);
}
__device__ __forceinline__ float silu_f(float x) {
    return x / (1.0f + __expf(-x));
}

// ---------------- LayerNorm: h = LN(x)*gamma+beta, stored bf16 ----------------
__global__ void ln_kernel(const float* __restrict__ x,
                          const float* __restrict__ gamma,
                          const float* __restrict__ beta,
                          unsigned short* __restrict__ hb) {
    const int node = blockIdx.x;
    const int t = threadIdx.x;            // 128 threads
    float v = x[node * HIDDEN + t];
    __shared__ float red[HIDDEN];
    red[t] = v;
    __syncthreads();
    #pragma unroll
    for (int s = 64; s > 0; s >>= 1) { if (t < s) red[t] += red[t + s]; __syncthreads(); }
    float mu = red[0] * (1.0f / HIDDEN);
    __syncthreads();
    float d = v - mu;
    red[t] = d * d;
    __syncthreads();
    #pragma unroll
    for (int s = 64; s > 0; s >>= 1) { if (t < s) red[t] += red[t + s]; __syncthreads(); }
    float var = red[0] * (1.0f / HIDDEN);
    float hn = d * rsqrtf(var + LN_EPS) * gamma[t] + beta[t];
    hb[node * HIDDEN + t] = f2bf(hn);
}

// ---------------- lattice inner products: ips[g,i,k] = <L[g,i,:],L[g,k,:]> ----
__global__ void lat_kernel(const float* __restrict__ L, float* __restrict__ ips, int G) {
    int idx = blockIdx.x * blockDim.x + threadIdx.x;
    if (idx >= G * 9) return;
    int g = idx / 9, r = idx % 9, i = r / 3, k = r % 3;
    const float* Lg = L + g * 9;
    ips[idx] = Lg[i*3+0]*Lg[k*3+0] + Lg[i*3+1]*Lg[k*3+1] + Lg[i*3+2]*Lg[k*3+2];
}

// ---- pack weights f32 -> bf16, swizzled to exact per-lane B-fragment order ----
// layout: idx = ((kk*8 + w)*32 + l)*8 + v
//   lane l: half=(l>>4), nl=l&15 ; fragment dword v holds K rows (kb, kb+1) at
//   kb = kk*32 + (v>>2)*16 + half*8 + (v&3)*2, column col = w*16 + nl.
// Each lane's 8 dwords are contiguous (32B) -> loads as 2x global_load_b128.
__global__ void pack_kernel(const float* __restrict__ eW1, const float* __restrict__ eW2,
                            const float* __restrict__ nW1, const float* __restrict__ nW2,
                            unsigned* __restrict__ eW1p, unsigned* __restrict__ eW2p,
                            unsigned* __restrict__ nW1p, unsigned* __restrict__ nW2p) {
    int idx = blockIdx.x * blockDim.x + threadIdx.x;
    int v  = idx & 7;
    int l  = (idx >> 3) & 31;
    int w  = (idx >> 8) & 7;
    int kk = idx >> 11;
    int half = (l >> 4) & 1, nl = l & 15;
    int col = w * 16 + nl;
    int kb  = kk * 32 + (v >> 2) * 16 + half * 8 + (v & 3) * 2;
    if (idx < (K1PAD / 32) * 2048) {      // 9 K-steps
        float a = (kb     < EDGE_IN) ? eW1[kb       * HIDDEN + col] : 0.0f;
        float b = (kb + 1 < EDGE_IN) ? eW1[(kb + 1) * HIDDEN + col] : 0.0f;
        eW1p[idx] = pack2(a, b);
    }
    if (idx < 4 * 2048) {                 // K=128
        eW2p[idx] = pack2(eW2[kb * HIDDEN + col], eW2[(kb + 1) * HIDDEN + col]);
        nW2p[idx] = pack2(nW2[kb * HIDDEN + col], nW2[(kb + 1) * HIDDEN + col]);
    }
    if (idx < 8 * 2048) {                 // K=256
        nW1p[idx] = pack2(nW1[kb * HIDDEN + col], nW1[(kb + 1) * HIDDEN + col]);
    }
}

// ---------------- zero accumulators ----------------
__global__ void zero_kernel(float* __restrict__ agg, float* __restrict__ counts, int N) {
    int idx = blockIdx.x * blockDim.x + threadIdx.x;
    if (idx < N * HIDDEN) agg[idx] = 0.0f;
    if (idx < N) counts[idx] = 0.0f;
}

// ---------------- per-source-node edge counts ----------------
__global__ void count_kernel(const int* __restrict__ eidx, float* __restrict__ counts, int E) {
    int e = blockIdx.x * blockDim.x + threadIdx.x;
    if (e < E)
        __hip_atomic_fetch_add(&counts[eidx[e]], 1.0f, __ATOMIC_RELAXED, __HIP_MEMORY_SCOPE_AGENT);
}

// ---------------- edge MLP + scatter-add (WMMA bf16) ----------------
__global__ void __launch_bounds__(256)
edge_kernel(const unsigned short* __restrict__ hb,
            const float* __restrict__ frac,
            const float* __restrict__ lat,
            const int* __restrict__ eidx,     // [2][E]
            const int* __restrict__ e2g,
            const unsigned* __restrict__ W1p, const float* __restrict__ b1,
            const unsigned* __restrict__ W2p, const float* __restrict__ b2,
            float* __restrict__ agg, int E) {
    __shared__ unsigned short sIn[16 * K1PAD];   // 16 edges x 288 bf16 (row = 36 v4u)
    __shared__ unsigned short sMid[16 * HIDDEN]; // 16 edges x 128 bf16
    __shared__ int sSrc[16], sDst[16], sG[16];

    const int tid = threadIdx.x;
    const int base = blockIdx.x * 16;
    if (tid < 16) {
        int e = base + tid; if (e >= E) e = E - 1;
        sSrc[tid] = eidx[e];
        sDst[tid] = eidx[E + e];
        sG[tid]   = e2g[e];
    }
    __syncthreads();

    // --- build tile: h rows as 16B-chunk copies (b128), tail scalar ---
    {
        const v4u* hb4 = (const v4u*)hb;     // 16 v4u per node row
        v4u* sIn4 = (v4u*)sIn;               // row stride 36 v4u
        for (int t = tid; t < 16 * 32; t += 256) {   // 16 edges x (hi:16 + hj:16) chunks
            int e = t >> 5, rem = t & 31;
            int which = rem >> 4, c = rem & 15;
            int node = which ? sDst[e] : sSrc[e];
            sIn4[e * 36 + rem] = hb4[node * 16 + c];
        }
        for (int t = tid; t < 16 * 32; t += 256) {   // k = 256..287 tail
            int e = t >> 5, k = 256 + (t & 31);
            float val = 0.0f;
            if (k < 2 * HIDDEN + 3) {
                int c = k - 2 * HIDDEN;
                float d = frac[sDst[e] * 3 + c] - frac[sSrc[e] * 3 + c];
                val = d - floorf(d);                 // (x) mod 1.0
            } else if (k < 2 * HIDDEN + 12) {
                val = lat[sG[e] * 9 + (k - (2 * HIDDEN + 3))];
            }
            sIn[e * K1PAD + k] = f2bf(val);
        }
    }
    __syncthreads();

    const int w = tid >> 5, l = tid & 31;
    const int half = (l >> 4) & 1;       // which 16-lane group
    const int n = l & 15;                // N col / A row (m)
    const int nb = w * 16;               // this wave's 16-column slab
    const unsigned* sInU = (const unsigned*)sIn;
    const unsigned fragBase = (unsigned)(w * 256 + l * 8);   // per-lane fragment offset

    // GEMM1: [16 x 288] x [288 x 128], 9 K-steps
    v8f c = {};
    #pragma unroll
    for (int kk = 0; kk < K1PAD / 32; ++kk) {
        v8u au;
        v8u bu = *(const v8u*)(W1p + fragBase + kk * 2048);
        #pragma unroll
        for (int v = 0; v < 8; ++v) {
            int ka = kk * 32 + (v >> 2) * 16 + half * 8 + (v & 3) * 2;   // A layout (ISA 7.12.2)
            au[v] = sInU[n * (K1PAD / 2) + (ka >> 1)];
        }
        c = __builtin_amdgcn_wmma_f32_16x16x32_bf16(false, __builtin_bit_cast(v16bf, au),
                                                    false, __builtin_bit_cast(v16bf, bu),
                                                    (short)0, c, false, false);
    }
    // bias + SiLU -> LDS bf16
    #pragma unroll
    for (int v = 0; v < 8; ++v) {
        int m = v + half * 8;
        sMid[m * HIDDEN + nb + n] = f2bf(silu_f(c[v] + b1[nb + n]));
    }
    __syncthreads();

    // GEMM2: [16 x 128] x [128 x 128], 4 K-steps
    const unsigned* sMidU = (const unsigned*)sMid;
    v8f c2 = {};
    #pragma unroll
    for (int kk = 0; kk < HIDDEN / 32; ++kk) {
        v8u au;
        v8u bu = *(const v8u*)(W2p + fragBase + kk * 2048);
        #pragma unroll
        for (int v = 0; v < 8; ++v) {
            int ka = kk * 32 + (v >> 2) * 16 + half * 8 + (v & 3) * 2;
            au[v] = sMidU[n * (HIDDEN / 2) + (ka >> 1)];
        }
        c2 = __builtin_amdgcn_wmma_f32_16x16x32_bf16(false, __builtin_bit_cast(v16bf, au),
                                                     false, __builtin_bit_cast(v16bf, bu),
                                                     (short)0, c2, false, false);
    }
    // bias + SiLU + scatter-add by src
    #pragma unroll
    for (int v = 0; v < 8; ++v) {
        int m = v + half * 8;
        if (base + m < E) {
            float x = silu_f(c2[v] + b2[nb + n]);
            __hip_atomic_fetch_add(&agg[sSrc[m] * HIDDEN + nb + n], x,
                                   __ATOMIC_RELAXED, __HIP_MEMORY_SCOPE_AGENT);
        }
    }
}

// ---------------- node MLP + residual (WMMA bf16) ----------------
__global__ void __launch_bounds__(256)
node_kernel(const unsigned short* __restrict__ hb,
            const float* __restrict__ agg,
            const float* __restrict__ counts,
            const unsigned* __restrict__ W1p, const float* __restrict__ b1,
            const unsigned* __restrict__ W2p, const float* __restrict__ b2,
            const float* __restrict__ xin,
            float* __restrict__ out, int N) {
    __shared__ unsigned short sIn[16 * 256];     // 16 nodes x concat(h, agg/cnt); row = 32 v4u
    __shared__ unsigned short sMid[16 * HIDDEN];
    const int tid = threadIdx.x;
    const int base = blockIdx.x * 16;

    // h part: b128 chunk copies
    {
        const v4u* hb4 = (const v4u*)hb;
        v4u* sIn4 = (v4u*)sIn;
        for (int t = tid; t < 16 * 16; t += 256) {
            int e = t >> 4, cidx = t & 15;
            int node = base + e; if (node >= N) node = N - 1;
            sIn4[e * 32 + cidx] = hb4[node * 16 + cidx];
        }
    }
    // agg/cnt part: two f32 -> one packed u32 per thread-iter
    {
        unsigned* sInW = (unsigned*)sIn;         // row stride 128 u32, agg at +64
        for (int t = tid; t < 16 * 64; t += 256) {
            int e = t >> 6, kp = t & 63;
            int node = base + e; if (node >= N) node = N - 1;
            float cnt = counts[node];
            float inv = 1.0f / (cnt > 1.0f ? cnt : 1.0f);
            sInW[e * 128 + 64 + kp] = pack2(agg[node * HIDDEN + 2 * kp] * inv,
                                            agg[node * HIDDEN + 2 * kp + 1] * inv);
        }
    }
    __syncthreads();

    const int w = tid >> 5, l = tid & 31;
    const int half = (l >> 4) & 1;
    const int n = l & 15;
    const int nb = w * 16;
    const unsigned* sInU = (const unsigned*)sIn;
    const unsigned fragBase = (unsigned)(w * 256 + l * 8);

    // GEMM1: [16 x 256] x [256 x 128], 8 K-steps
    v8f c = {};
    #pragma unroll
    for (int kk = 0; kk < 256 / 32; ++kk) {
        v8u au;
        v8u bu = *(const v8u*)(W1p + fragBase + kk * 2048);
        #pragma unroll
        for (int v = 0; v < 8; ++v) {
            int ka = kk * 32 + (v >> 2) * 16 + half * 8 + (v & 3) * 2;
            au[v] = sInU[n * 128 + (ka >> 1)];
        }
        c = __builtin_amdgcn_wmma_f32_16x16x32_bf16(false, __builtin_bit_cast(v16bf, au),
                                                    false, __builtin_bit_cast(v16bf, bu),
                                                    (short)0, c, false, false);
    }
    #pragma unroll
    for (int v = 0; v < 8; ++v) {
        int m = v + half * 8;
        sMid[m * HIDDEN + nb + n] = f2bf(silu_f(c[v] + b1[nb + n]));
    }
    __syncthreads();

    // GEMM2: [16 x 128] x [128 x 128], 4 K-steps
    const unsigned* sMidU = (const unsigned*)sMid;
    v8f c2 = {};
    #pragma unroll
    for (int kk = 0; kk < HIDDEN / 32; ++kk) {
        v8u au;
        v8u bu = *(const v8u*)(W2p + fragBase + kk * 2048);
        #pragma unroll
        for (int v = 0; v < 8; ++v) {
            int ka = kk * 32 + (v >> 2) * 16 + half * 8 + (v & 3) * 2;
            au[v] = sMidU[n * (HIDDEN / 2) + (ka >> 1)];
        }
        c2 = __builtin_amdgcn_wmma_f32_16x16x32_bf16(false, __builtin_bit_cast(v16bf, au),
                                                     false, __builtin_bit_cast(v16bf, bu),
                                                     (short)0, c2, false, false);
    }
    #pragma unroll
    for (int v = 0; v < 8; ++v) {
        int m = v + half * 8;
        int node = base + m;
        if (node < N) {
            float x = silu_f(c2[v] + b2[nb + n]);
            out[node * HIDDEN + nb + n] = xin[node * HIDDEN + nb + n] + x;
        }
    }
}

extern "C" void kernel_launch(void* const* d_in, const int* in_sizes, int n_in,
                              void* d_out, int out_size, void* d_ws, size_t ws_size,
                              hipStream_t stream) {
    const float* node_features = (const float*)d_in[0];
    const float* frac          = (const float*)d_in[1];
    const float* lattices      = (const float*)d_in[2];
    const int*   eidx          = (const int*)d_in[3];
    const int*   e2g           = (const int*)d_in[4];
    /* d_in[5] num_atoms: unused */
    const float* gamma = (const float*)d_in[6];
    const float* beta  = (const float*)d_in[7];
    const float* eW1 = (const float*)d_in[8];
    const float* eb1 = (const float*)d_in[9];
    const float* eW2 = (const float*)d_in[10];
    const float* eb2 = (const float*)d_in[11];
    const float* nW1 = (const float*)d_in[12];
    const float* nb1 = (const float*)d_in[13];
    const float* nW2 = (const float*)d_in[14];
    const float* nb2 = (const float*)d_in[15];

    const int N = in_sizes[0] / HIDDEN;
    const int E = in_sizes[4];
    const int G = in_sizes[5];
    float* out = (float*)d_out;

    // workspace carve-out (256B aligned)
    char* ws = (char*)d_ws;
    size_t off = 0;
    auto carve = [&](size_t bytes) -> void* {
        void* p = ws + off;
        off += (bytes + 255) & ~(size_t)255;
        return p;
    };
    unsigned short* hb   = (unsigned short*)carve((size_t)N * HIDDEN * 2);
    float* agg           = (float*)carve((size_t)N * HIDDEN * 4);
    float* counts        = (float*)carve((size_t)N * 4);
    float* lat           = (float*)carve((size_t)G * 9 * 4);
    unsigned* eW1p       = (unsigned*)carve((size_t)(K1PAD / 32) * 2048 * 4);
    unsigned* eW2p       = (unsigned*)carve((size_t)4 * 2048 * 4);
    unsigned* nW1p       = (unsigned*)carve((size_t)8 * 2048 * 4);
    unsigned* nW2p       = (unsigned*)carve((size_t)4 * 2048 * 4);
    (void)ws_size; (void)n_in; (void)out_size;

    ln_kernel<<<N, HIDDEN, 0, stream>>>(node_features, gamma, beta, hb);
    lat_kernel<<<(G * 9 + 255) / 256, 256, 0, stream>>>(lattices, lat, G);
    pack_kernel<<<((K1PAD / 32) * 2048 + 255) / 256, 256, 0, stream>>>(
        eW1, eW2, nW1, nW2, eW1p, eW2p, nW1p, nW2p);
    zero_kernel<<<((size_t)N * HIDDEN + 255) / 256, 256, 0, stream>>>(agg, counts, N);
    count_kernel<<<(E + 255) / 256, 256, 0, stream>>>(eidx, counts, E);
    edge_kernel<<<(E + 15) / 16, 256, 0, stream>>>(hb, frac, lat, eidx, e2g,
                                                   eW1p, eb1, eW2p, eb2, agg, E);
    node_kernel<<<(N + 15) / 16, 256, 0, stream>>>(hb, agg, counts,
                                                   nW1p, nb1, nW2p, nb2,
                                                   node_features, out, N);
}
